// RevRNN_76347338653736
// MI455X (gfx1250) — compile-verified, matrix operand order checked by reading, not run on hardware
//
#include <hip/hip_runtime.h>
#include <hip/hip_bf16.h>
#include <math.h>

// Problem constants (from reference setup_inputs)
#define BATCH 512
#define SEQ   256
#define FEAT  256
#define HID   2048
#define HH    1024          // HID/2
#define KDIM  1282          // HH + FEAT + 2
#define KT    41            // ceil(KDIM/32)
#define KP    (KT * 32)     // 1312 padded K
#define TSTEPS 264          // SEQ + DELAY(8)

typedef __attribute__((ext_vector_type(16))) __bf16 v16bf;
typedef __attribute__((ext_vector_type(8)))  float  v8f;
typedef int v4i __attribute__((vector_size(4 * sizeof(int))));

#define AS1 __attribute__((address_space(1)))
#define AS3 __attribute__((address_space(3)))

// Async global->LDS staging (CDNA5 GLOBAL_LOAD_ASYNC_TO_LDS_B128, ASYNCcnt),
// feature-guarded; falls back to load + ds_store staging.
#if defined(__has_builtin)
#if __has_builtin(__builtin_amdgcn_global_load_async_to_lds_b128)
#define USE_ASYNC_LDS 1
#endif
#endif
#ifndef USE_ASYNC_LDS
#define USE_ASYNC_LDS 0
#endif

static __device__ __forceinline__ void cp16(const void* g, void* l) {
#if USE_ASYNC_LDS
  __builtin_amdgcn_global_load_async_to_lds_b128((AS1 v4i*)g, (AS3 v4i*)l, 0, 0);
#else
  *(int4*)l = *(const int4*)g;
#endif
}

static __device__ __forceinline__ void stage_wait() {
#if USE_ASYNC_LDS
#if __has_builtin(__builtin_amdgcn_s_wait_asynccnt)
  __builtin_amdgcn_s_wait_asynccnt(0);
#else
  asm volatile("s_wait_asynccnt 0x0" ::: "memory");
#endif
#endif
  __syncthreads();
}

// ---------------------------------------------------------------------------
// Pack fp32 weight W[HID, KDIM] into bf16 B-matrix tiles:
//   Bp[(kt*HID + n)*32 + kk] = W[n, kt*32+kk]   (zero for k >= KDIM)
// Each WMMA B-fragment lane then loads 16 contiguous bf16 (32 bytes).
// ---------------------------------------------------------------------------
__global__ __launch_bounds__(256) void packW_kernel(const float* __restrict__ W,
                                                    __bf16* __restrict__ Bp) {
  size_t idx = (size_t)blockIdx.x * 256 + threadIdx.x;   // over KT*HID*32
  int kk = (int)(idx & 31);
  int n  = (int)((idx >> 5) & (HID - 1));
  int kt = (int)(idx >> 16);                             // HID*32 == 65536
  int k  = kt * 32 + kk;
  float v = (k < KDIM) ? W[(size_t)n * KDIM + k] : 0.0f;
  Bp[idx] = (__bf16)v;
}

// ---------------------------------------------------------------------------
// Split initial hidden state [B, HID] into state0/state1 [B, HH] fp32
// ---------------------------------------------------------------------------
__global__ __launch_bounds__(256) void init_kernel(const float* __restrict__ hidden,
                                                   float* __restrict__ st0,
                                                   float* __restrict__ st1) {
  size_t idx = (size_t)blockIdx.x * 256 + threadIdx.x;   // over BATCH*HID
  int r = (int)(idx >> 11);
  int c = (int)(idx & (HID - 1));
  float v = hidden[idx];
  if (c < HH) st0[(size_t)r * HH + c] = v;
  else        st1[(size_t)r * HH + (c - HH)] = v;
}

// ---------------------------------------------------------------------------
// Fused per-half-step prep:
//   optional gate update:  state = O[:, :HH]*sig(O[:, HH:]) + state*(1-sig)
//   batch stats (mean, unbiased std over B) via LDS reduction
//   mish( (x-mean)/(std+1e-6)*w + b )  ->  bf16, packed in WMMA A-tile layout
//     Ap[((r/16)*KT + k/32)*512 + (r%16)*32 + (k%32)]
// Blocks 0..15: 64 columns each, 4 row-segments x 64 cols = 256 threads.
// Block 16 (only when grid=17): fills k = HH..KP-1 (x_t, pos, zero pad).
// ---------------------------------------------------------------------------
__global__ __launch_bounds__(256) void prep_kernel(
    float* __restrict__ state, const float* __restrict__ O,
    const float* __restrict__ fnInput,
    const float* __restrict__ bnW, const float* __restrict__ bnB,
    __bf16* __restrict__ Ap, int t, int doUpdate) {
  if (blockIdx.x < 16) {
    int ci   = threadIdx.x & 63;
    int c    = blockIdx.x * 64 + ci;
    int rseg = threadIdx.x >> 6;
    int r0   = rseg * 128;

    float sum = 0.0f, sq = 0.0f;
    for (int r = r0; r < r0 + 128; ++r) {
      float v;
      if (doUpdate) {
        float o0 = O[(size_t)r * HID + c];
        float o1 = O[(size_t)r * HID + HH + c];
        float g  = 1.0f / (1.0f + __expf(-o1));
        v = o0 * g + state[(size_t)r * HH + c] * (1.0f - g);
        state[(size_t)r * HH + c] = v;
      } else {
        v = state[(size_t)r * HH + c];
      }
      sum += v;
      sq  += v * v;
    }

    __shared__ float sSum[4][64];
    __shared__ float sSq[4][64];
    __shared__ float sMean[64];
    __shared__ float sScale[64];
    __shared__ float sBias[64];
    sSum[rseg][ci] = sum;
    sSq[rseg][ci]  = sq;
    __syncthreads();
    if (rseg == 0) {
      float s = sSum[0][ci] + sSum[1][ci] + sSum[2][ci] + sSum[3][ci];
      float q = sSq[0][ci]  + sSq[1][ci]  + sSq[2][ci]  + sSq[3][ci];
      float mean = s * (1.0f / (float)BATCH);
      float var  = (q - (float)BATCH * mean * mean) * (1.0f / (float)(BATCH - 1));
      float sd   = sqrtf(fmaxf(var, 0.0f));
      sMean[ci]  = mean;
      sScale[ci] = bnW[c] / (sd + 1e-6f);
      sBias[ci]  = bnB[c];
    }
    __syncthreads();
    float mean = sMean[ci], scale = sScale[ci], bias = sBias[ci];

    int kt = c >> 5, kk = c & 31;
    for (int r = r0; r < r0 + 128; ++r) {
      float v  = state[(size_t)r * HH + c];
      float o  = (v - mean) * scale + bias;
      float sp = (o > 20.0f) ? o : log1pf(__expf(o));   // softplus
      float m  = o * tanhf(sp);                          // mish
      Ap[((size_t)(r >> 4) * KT + kt) * 512 + (size_t)(r & 15) * 32 + kk] = (__bf16)m;
    }
  } else {
    // sequence input + position + zero pad columns (k = HH .. KP-1)
    float idxv   = (float)(t + 1);
    float factor = (float)(TSTEPS + 1) * 0.5f;
    for (int c = threadIdx.x; c < KP - HH; c += 256) {
      int k  = HH + c;
      int kt = k >> 5, kk = k & 31;
      for (int r = 0; r < BATCH; ++r) {
        float v = 0.0f;
        if (c < FEAT) {
          v = (t < SEQ) ? fnInput[((size_t)r * SEQ + t) * FEAT + c] : 0.0f;
        } else if (c == FEAT) {
          v = idxv;
        } else if (c == FEAT + 1) {
          v = (idxv - factor) / factor;
        }
        Ap[((size_t)(r >> 4) * KT + kt) * 512 + (size_t)(r & 15) * 32 + kk] = (__bf16)v;
      }
    }
  }
}

// ---------------------------------------------------------------------------
// GEMM: O[BATCH, HID] (fp32) = A[BATCH, KP] (bf16) x Bp[KP, HID] (bf16)
// WG = 256 threads (8 waves), WG tile 128(M) x 128(N), grid (16, 4).
// Per K-slice (K=32): stage A-tile (8x1KB) + B-tile (8KB contiguous) into LDS
// with async global->LDS copies, double buffered; every wave then ds_loads its
// fragments (shared across waves) and issues 8 v_wmma_f32_16x16x32_bf16.
// Wave w: M-tiles {2*(w&3), 2*(w&3)+1}, N-tiles {4*(w>>2) .. +3}.
// ---------------------------------------------------------------------------
__global__ __launch_bounds__(256) void gemm_kernel(
    const __bf16* __restrict__ Ap, const __bf16* __restrict__ Bp,
    float* __restrict__ O) {
  __shared__ __align__(32) char As[2][8192];   // 8 M-tiles x [16][32] bf16
  __shared__ __align__(32) char Bs[2][8192];   // 8 N-tiles x [16][32] bf16

  const int tid  = threadIdx.x;
  const int wave = tid >> 5;
  const int lane = tid & 31;
  const int lr = lane & 15;      // row (A) / column (B) within 16x16 tile
  const int lh = lane >> 4;      // K-half select
  const int mg = wave & 3;       // M group -> tiles 2*mg, 2*mg+1
  const int ng = wave >> 2;      // N group -> tiles 4*ng .. 4*ng+3
  const int mtBase = blockIdx.y * 8;      // first M tile of WG
  const int nBase  = blockIdx.x * 128;    // first N column of WG

  // staging thread mapping
  const int ti = tid >> 5;       // A tile handled by this thread
  const int li = tid & 31;       // 32B chunk within A tile
  const char* aG = (const char*)(Ap) + ((size_t)(mtBase + ti) * KT) * 1024
                 + (size_t)li * 32;                       // + kt*1024
  const char* bG = (const char*)(Bp) + ((size_t)nBase) * 64
                 + (size_t)tid * 16;                      // + kt*131072 (+4096)

  v8f acc[2][4] = {};

  // stage slice 0 into buffer 0
  {
    const char* a = aG;
    const char* b = bG;
    cp16(a,        &As[0][ti * 1024 + li * 32]);
    cp16(a + 16,   &As[0][ti * 1024 + li * 32 + 16]);
    cp16(b,        &Bs[0][tid * 16]);
    cp16(b + 4096, &Bs[0][tid * 16 + 4096]);
  }
  stage_wait();

  int buf = 0;
  for (int kt = 0; kt < KT; ++kt) {
    if (kt + 1 < KT) {           // issue async stage of next slice (overlaps WMMA)
      const char* a = aG + (size_t)(kt + 1) * 1024;
      const char* b = bG + (size_t)(kt + 1) * 131072;
      cp16(a,        &As[buf ^ 1][ti * 1024 + li * 32]);
      cp16(a + 16,   &As[buf ^ 1][ti * 1024 + li * 32 + 16]);
      cp16(b,        &Bs[buf ^ 1][tid * 16]);
      cp16(b + 4096, &Bs[buf ^ 1][tid * 16 + 4096]);
    }

    const char* aT = As[buf];
    const char* bT = Bs[buf];
    v16bf a0 = *(const v16bf*)(aT + (2 * mg + 0) * 1024 + lr * 64 + lh * 32);
    v16bf a1 = *(const v16bf*)(aT + (2 * mg + 1) * 1024 + lr * 64 + lh * 32);
#pragma unroll
    for (int j = 0; j < 4; ++j) {
      v16bf b = *(const v16bf*)(bT + (4 * ng + j) * 1024 + lr * 64 + lh * 32);
      acc[0][j] = __builtin_amdgcn_wmma_f32_16x16x32_bf16(
          false, a0, false, b, (short)0, acc[0][j], false, false);
      acc[1][j] = __builtin_amdgcn_wmma_f32_16x16x32_bf16(
          false, a1, false, b, (short)0, acc[1][j], false, false);
    }

    if (kt + 1 < KT) stage_wait();   // next buffer ready; everyone done reading
    buf ^= 1;
  }

  // C/D layout: VGPR v, lanes 0-15 -> M=v, lanes 16-31 -> M=8+v, N=lane%16
#pragma unroll
  for (int i = 0; i < 2; ++i) {
    int rowBase = (mtBase + 2 * mg + i) * 16 + lh * 8;
#pragma unroll
    for (int j = 0; j < 4; ++j) {
      int col = nBase + (4 * ng + j) * 16 + lr;
#pragma unroll
      for (int v = 0; v < 8; ++v) {
        O[(size_t)(rowBase + v) * HID + col] = acc[i][j][v];
      }
    }
  }
}

// ---------------------------------------------------------------------------
// Final: apply last gate update (from O of lin1 at t=T-1) to state0 and emit
// d_out[B, HID] = concat(state0_new, state1)
// ---------------------------------------------------------------------------
__global__ __launch_bounds__(256) void finish_kernel(const float* __restrict__ O,
                                                     const float* __restrict__ st0,
                                                     const float* __restrict__ st1,
                                                     float* __restrict__ out) {
  size_t idx = (size_t)blockIdx.x * 256 + threadIdx.x;   // over BATCH*HH
  int r = (int)(idx >> 10);
  int c = (int)(idx & (HH - 1));
  float o0 = O[(size_t)r * HID + c];
  float o1 = O[(size_t)r * HID + HH + c];
  float g  = 1.0f / (1.0f + __expf(-o1));
  out[(size_t)r * HID + c]      = o0 * g + st0[idx] * (1.0f - g);
  out[(size_t)r * HID + HH + c] = st1[idx];
}

// ---------------------------------------------------------------------------
extern "C" void kernel_launch(void* const* d_in, const int* in_sizes, int n_in,
                              void* d_out, int out_size, void* d_ws, size_t ws_size,
                              hipStream_t stream) {
  (void)in_sizes; (void)n_in; (void)out_size; (void)ws_size;
  const float* fnInput = (const float*)d_in[0];   // [B, S, F]
  const float* hidden  = (const float*)d_in[1];   // [B, HID]
  const float* lin0    = (const float*)d_in[2];   // [HID, KDIM]
  const float* lin1    = (const float*)d_in[3];   // [HID, KDIM]
  const float* bnW0    = (const float*)d_in[4];
  const float* bnB0    = (const float*)d_in[5];
  const float* bnW1    = (const float*)d_in[6];
  const float* bnB1    = (const float*)d_in[7];
  float* out = (float*)d_out;

  // Workspace layout (all 256B aligned)
  char* ws = (char*)d_ws;
  const size_t szB = (size_t)KT * HID * 32 * sizeof(__bf16);   // 5,373,952 B each
  const size_t szA = (size_t)BATCH * KP * sizeof(__bf16);      // 1,343,488 B
  const size_t szO = (size_t)BATCH * HID * sizeof(float);      // 4 MB
  const size_t szS = (size_t)BATCH * HH * sizeof(float);       // 2 MB each
  __bf16* Bp0 = (__bf16*)(ws);
  __bf16* Bp1 = (__bf16*)(ws + szB);
  __bf16* Ap  = (__bf16*)(ws + 2 * szB);
  float*  O   = (float*)(ws + 2 * szB + szA);
  float*  st0 = (float*)(ws + 2 * szB + szA + szO);
  float*  st1 = (float*)(ws + 2 * szB + szA + szO + szS);

  // One-time (per launch) weight packing fp32 -> bf16 WMMA B layout
  int packBlocks = (int)(((size_t)KT * HID * 32) / 256);       // 10496
  packW_kernel<<<packBlocks, 256, 0, stream>>>(lin0, Bp0);
  packW_kernel<<<packBlocks, 256, 0, stream>>>(lin1, Bp1);
  init_kernel<<<(BATCH * HID) / 256, 256, 0, stream>>>(hidden, st0, st1);

  for (int t = 0; t < TSTEPS; ++t) {
    // half-step 1: finish previous state0 update, normalize+mish+pack, GEMM lin0
    prep_kernel<<<17, 256, 0, stream>>>(st0, O, fnInput, bnW0, bnB0, Ap, t,
                                        (t > 0) ? 1 : 0);
    gemm_kernel<<<dim3(16, 4), 256, 0, stream>>>(Ap, Bp0, O);
    // half-step 2: update state1 from O1, normalize+mish+pack, GEMM lin1
    prep_kernel<<<16, 256, 0, stream>>>(st1, O, fnInput, bnW1, bnB1, Ap, t, 1);
    gemm_kernel<<<dim3(16, 4), 256, 0, stream>>>(Ap, Bp1, O);
  }

  finish_kernel<<<(BATCH * HH) / 256, 256, 0, stream>>>(O, st0, st1, out);
}
